// _OutputModule_15951508537659
// MI455X (gfx1250) — matrix operand order repaired and checked by reading, not compiled, not run on hardware
//
#include <hip/hip_runtime.h>
#include <hip/hip_bf16.h>

typedef __attribute__((ext_vector_type(16))) __bf16 v16bf;
typedef __attribute__((ext_vector_type(2)))  __bf16 v2bf;
typedef __attribute__((ext_vector_type(8)))  float  v8f;

// ---------- fast device ops (gfx1250 native where available) ----------
__device__ __forceinline__ float fast_tanh(float x) {
#if __has_builtin(__builtin_amdgcn_tanhf)
    return __builtin_amdgcn_tanhf(x);        // v_tanh_f32 (gfx1250 TRANS op)
#else
    return tanhf(x);
#endif
}

__device__ __forceinline__ unsigned short f2bf(float f) {
    __bf16 h = (__bf16)f;                    // hardware f32->bf16 RNE convert
    return __builtin_bit_cast(unsigned short, h);
}
__device__ __forceinline__ float bf2f(unsigned short h) {
    return __uint_as_float(((unsigned int)h) << 16);
}
__device__ __forceinline__ unsigned int pack_bf16(float a, float b) {
#if __has_builtin(__builtin_amdgcn_cvt_pk_bf16_f32)
    v2bf r = __builtin_amdgcn_cvt_pk_bf16_f32(a, b);   // v_cvt_pk_bf16_f32
    return __builtin_bit_cast(unsigned int, r);
#else
    return (unsigned int)f2bf(a) | ((unsigned int)f2bf(b) << 16);
#endif
}

// ---------- WMMA wrapper ----------
__device__ __forceinline__ v8f wmma_bf16(v16bf a, v16bf b, v8f c) {
    return __builtin_amdgcn_wmma_f32_16x16x32_bf16(false, a, false, b, (short)0, c, false, false);
}

// A-operand (16x32 bf16, M x K): lane<16 -> M=lane, K=[k0, k0+16)
//                                lane>=16 -> same M, K=[k0+16, k0+32).
// src is row-major [rows][128] bf16 (LDS).
__device__ __forceinline__ v16bf load_a_rows(const unsigned short* __restrict__ src,
                                             int row, int k0, int half) {
    union { v16bf v; uint4 q[2]; } u;
    u.q[0] = *(const uint4*)(src + row * 128 + k0 + half * 16);
    u.q[1] = *(const uint4*)(src + row * 128 + k0 + 8 + half * 16);
    return u.v;
}

// B-operand (32x16 bf16, K x N): lane<16 -> N=lane, K=[k0,k0+16); lane>=16 -> N=lane-16, K=[k0+16,k0+32).
// Tbf is the transposed matrix [N-col][K] bf16, K contiguous (global, L2-hot).
__device__ __forceinline__ v16bf load_b_tbf(const unsigned short* __restrict__ Tbf,
                                            int col, int k0, int half) {
    union { v16bf v; uint4 q[2]; } u;
    const uint4* p = (const uint4*)(Tbf + col * 128 + k0 + half * 16);
    u.q[0] = p[0];
    u.q[1] = p[1];
    return u.v;
}

// ================= small preparation kernels =================

// u[b,c] = sum_q emb[queries[b,q], c] * mult_mask[q, c]
__global__ __launch_bounds__(128) void compute_u(const int* __restrict__ queries,
                                                 const float* __restrict__ emb,
                                                 const float* __restrict__ mm,
                                                 float* __restrict__ u) {
    int b = blockIdx.x, c = threadIdx.x;
    float acc = 0.f;
    for (int q = 0; q < 20; ++q) {
        int t = queries[b * 20 + q];
        acc += emb[(size_t)t * 128 + c] * mm[q * 128 + c];
    }
    u[b * 128 + c] = acc;
}

// out[r,i] = sum_e x_r[e] * M[e,i];  x_r = emb[idx[r]] if idx else X[r]
__global__ __launch_bounds__(128) void gemv_rows(const float* __restrict__ X,
                                                 const int* __restrict__ idx,
                                                 const float* __restrict__ emb,
                                                 const float* __restrict__ M,
                                                 float* __restrict__ out) {
    int r = blockIdx.x, i = threadIdx.x;
    __shared__ float xs[128];
    const float* xr = idx ? (emb + (size_t)idx[r] * 128) : (X + r * 128);
    xs[i] = xr[i];
    __syncthreads();
    float acc = 0.f;
    for (int e = 0; e < 128; ++e) acc += xs[e] * M[e * 128 + i];
    out[r * 128 + i] = acc;
}

// T[i,e] = bf16(M[e,i])   (128x128 transpose + downconvert)
__global__ __launch_bounds__(128) void transpose_bf(const float* __restrict__ M,
                                                    unsigned short* __restrict__ T) {
    int i = blockIdx.x, e = threadIdx.x;
    T[i * 128 + e] = f2bf(M[e * 128 + i]);
}

// ================= main fused kernel =================
// Grid: (32 s-chunks of 16, 32 batches). Block: 128 threads = 4 waves.
// Phase 1 (per wave, 4 s each): stream memories[b,s] once, bf16 WMMA G = M@A (+kb+qc),
//   tanh, v-dot, softmax over NB=20, tm = attn^T @ M  -> tm_out + LDS bf16.
// Phase 2 (whole block): WMMA (tm@D + stories@E) for the 16 s, tanh/w-dot -> align2.
__global__ __launch_bounds__(128) void fused_intratemporal(
        const float* __restrict__ memories, const float* __restrict__ stories,
        const float* __restrict__ kb, const float* __restrict__ qc,
        const float* __restrict__ tf, const float* __restrict__ v,
        const float* __restrict__ w,
        const unsigned short* __restrict__ ATbf, const unsigned short* __restrict__ DTbf,
        const unsigned short* __restrict__ ETbf,
        float* __restrict__ tm_out, float* __restrict__ align2_out) {
    __shared__ __align__(16) unsigned short Mlds[4][32][128];  // per-wave memories tile (bf16, rows>=20 junk)
    __shared__ __align__(16) unsigned short TMbf[16][128];     // tm (bf16) for phase 2
    __shared__ __align__(16) unsigned short STbf[16][128];     // stories (bf16) for phase 2
    __shared__ float kb_s[20][128];
    __shared__ float qc_s[128], tf_s[128], v_s[128], w_s[128];
    __shared__ float align_s[4][32];
    __shared__ float align2_acc[16];

    const int tid  = threadIdx.x;
    const int lane = tid & 31;
    const int wid  = tid >> 5;
    const int half = lane >> 4;
    const int l15  = lane & 15;
    const int b      = blockIdx.y;
    const int s_base = blockIdx.x * 16;

    // phase 0: stage small per-batch vectors
    for (int idx = tid; idx < 20 * 128; idx += 128) ((float*)kb_s)[idx] = kb[idx];
    qc_s[tid] = qc[b * 128 + tid];
    tf_s[tid] = tf[b * 128 + tid];
    v_s[tid]  = v[tid];
    w_s[tid]  = w[tid];
    if (tid < 16) align2_acc[tid] = 0.f;
    __syncthreads();

    // rows handled by this lane in the second row tile: 16 + r + half*8.
    // Only the half==0 lane group (rows 16..19, r<4) is ever consumed after the
    // width-16 reduction; clamp the kb row index so half==1 lanes stay in bounds.
    const int row1_kb = (half == 0) ? 16 : 19;   // +r (r<4) below, clamped via min

    // ---------------- phase 1 ----------------
    for (int it = 0; it < 4; ++it) {
        const int s_local = it * 4 + wid;
        const int s = s_base + s_local;
        const float2* mp = (const float2*)(memories + ((size_t)(b * 512 + s) * 20) * 128);
        unsigned int* Mw = (unsigned int*)(&Mlds[wid][0][0]);
        #pragma unroll 8
        for (int t = 0; t < 40; ++t) {
            float2 x = mp[lane + 32 * t];
            Mw[lane + 32 * t] = pack_bf16(x.x, x.y);
        }
        if (it < 3)
            __builtin_prefetch(memories + ((size_t)(b * 512 + s + 4) * 20) * 128, 0, 1);

        float part[12];
        #pragma unroll
        for (int q = 0; q < 12; ++q) part[q] = 0.f;

        for (int ct = 0; ct < 8; ++ct) {
            const int i = ct * 16 + l15;
            const float vi  = v_s[i];
            const float qci = qc_s[i];
            v8f acc0 = {}, acc1 = {};
            #pragma unroll
            for (int ks = 0; ks < 4; ++ks) {
                const int k0 = ks * 32;
                v16bf bop = load_b_tbf(ATbf, i, k0, half);
                v16bf a0  = load_a_rows(&Mlds[wid][0][0], l15, k0, half);
                acc0 = wmma_bf16(a0, bop, acc0);
                v16bf a1  = load_a_rows(&Mlds[wid][0][0], 16 + l15, k0, half);
                acc1 = wmma_bf16(a1, bop, acc1);
            }
            #pragma unroll
            for (int r = 0; r < 8; ++r) {
                const int row0 = r + half * 8;            // 0..15 (<20 always)
                part[r] += vi * fast_tanh(acc0[r] + kb_s[row0][i] + qci);
            }
            #pragma unroll
            for (int r = 0; r < 4; ++r) {                 // rows 16..19 (half==0 group only)
                const int rk = (row1_kb + r > 19) ? 19 : (row1_kb + r);
                part[8 + r] += vi * fast_tanh(acc1[r] + kb_s[rk][i] + qci);
            }
        }
        // reduce each row's partial over its 16-lane half
        #pragma unroll
        for (int q = 0; q < 12; ++q) {
            float x = part[q];
            for (int m = 1; m < 16; m <<= 1) x += __shfl_xor(x, m, 16);
            part[q] = x;
        }
        if (l15 == 0) {
            #pragma unroll
            for (int r = 0; r < 8; ++r)
                align_s[wid][r + half * 8] = part[r];     // rows 0..15
            if (half == 0) {
                #pragma unroll
                for (int r = 0; r < 4; ++r)
                    align_s[wid][16 + r] = part[8 + r];   // rows 16..19
            }
        }
        // softmax over NB=20 (wave-local)
        float a  = (lane < 20) ? align_s[wid][lane] : -1e30f;
        float mx = a;
        for (int m = 16; m >= 1; m >>= 1) mx = fmaxf(mx, __shfl_xor(mx, m, 32));
        float e = (lane < 20) ? __expf(a - mx) : 0.f;
        float sm = e;
        for (int m = 16; m >= 1; m >>= 1) sm += __shfl_xor(sm, m, 32);
        if (lane < 20) align_s[wid][lane] = e / sm;

        // tm = attn^T @ M ; write f32 to global, bf16 to LDS; also stage stories bf16
        #pragma unroll
        for (int j = 0; j < 4; ++j) {
            const int c = lane + 32 * j;
            float acc = 0.f;
            #pragma unroll 4
            for (int n = 0; n < 20; ++n)
                acc += align_s[wid][n] * bf2f(Mlds[wid][n][c]);
            tm_out[((size_t)(b * 512 + s)) * 128 + c] = acc;
            TMbf[s_local][c] = f2bf(acc);
            STbf[s_local][c] = f2bf(stories[((size_t)(b * 512 + s)) * 128 + c]);
        }
    }
    __syncthreads();

    // ---------------- phase 2: align2 = sum_i w_i * tanh(tm@D + st@E + tf) ----------------
    float part2[8];
    #pragma unroll
    for (int r = 0; r < 8; ++r) part2[r] = 0.f;
    for (int tt = 0; tt < 2; ++tt) {
        const int ct = wid * 2 + tt;
        const int i  = ct * 16 + l15;
        v8f acc = {};
        #pragma unroll
        for (int ks = 0; ks < 4; ++ks) {
            const int k0 = ks * 32;
            v16bf bD = load_b_tbf(DTbf, i, k0, half);
            v16bf aT = load_a_rows(&TMbf[0][0], l15, k0, half);
            acc = wmma_bf16(aT, bD, acc);
            v16bf bE = load_b_tbf(ETbf, i, k0, half);
            v16bf aS = load_a_rows(&STbf[0][0], l15, k0, half);
            acc = wmma_bf16(aS, bE, acc);
        }
        const float wi = w_s[i], tfi = tf_s[i];
        #pragma unroll
        for (int r = 0; r < 8; ++r) part2[r] += wi * fast_tanh(acc[r] + tfi);
    }
    #pragma unroll
    for (int r = 0; r < 8; ++r) {
        float x = part2[r];
        for (int m = 1; m < 16; m <<= 1) x += __shfl_xor(x, m, 16);
        if (l15 == 0) atomicAdd(&align2_acc[r + half * 8], x);   // ds_add_f32
    }
    __syncthreads();
    if (tid < 16)
        align2_out[(size_t)b * 512 + s_base + tid] = align2_acc[tid];
}

// ================= temporal softmax over S + readout o =================
__global__ __launch_bounds__(256) void finalize_o(
        const float* __restrict__ align2, const float* __restrict__ mask,
        const float* __restrict__ tm, const float* __restrict__ uH,
        unsigned short* __restrict__ o_bf) {
    const int b = blockIdx.x, tid = threadIdx.x;
    const int lane = tid & 31, wid = tid >> 5;
    __shared__ float lg[512];
    __shared__ float red[8];
    __shared__ float stats[2];
    for (int s = tid; s < 512; s += 256) {
        float m = mask[b * 512 + s];
        float a = align2[b * 512 + s];
        lg[s] = (m != 0.f) ? a * m : -1e30f;
    }
    __syncthreads();
    float mx = fmaxf(lg[tid], lg[tid + 256]);
    for (int m = 16; m >= 1; m >>= 1) mx = fmaxf(mx, __shfl_xor(mx, m, 32));
    if (lane == 0) red[wid] = mx;
    __syncthreads();
    if (tid == 0) {
        float t = red[0];
        for (int q = 1; q < 8; ++q) t = fmaxf(t, red[q]);
        stats[0] = t;
    }
    __syncthreads();
    mx = stats[0];
    float e0 = __expf(lg[tid] - mx), e1 = __expf(lg[tid + 256] - mx);
    float sm = e0 + e1;
    __syncthreads();
    lg[tid] = e0; lg[tid + 256] = e1;
    for (int m = 16; m >= 1; m >>= 1) sm += __shfl_xor(sm, m, 32);
    if (lane == 0) red[wid] = sm;
    __syncthreads();
    if (tid == 0) {
        float t = 0.f;
        for (int q = 0; q < 8; ++q) t += red[q];
        stats[1] = 1.f / t;
    }
    __syncthreads();
    const float inv = stats[1];
    if (tid < 128) {
        float acc = 0.f;
        const float* tb = tm + ((size_t)b * 512) * 128 + tid;
        for (int s = 0; s < 512; ++s) acc += lg[s] * tb[(size_t)s * 128];
        acc = acc * inv + uH[b * 128 + tid];
        o_bf[b * 128 + tid] = f2bf(acc);
    }
}

// ================= logits = (o + uH) @ R  via bf16 WMMA =================
// M=32 (2 row tiles), N=40000 (2500 col tiles), K=128. R gathered f32->bf16 inline.
__global__ __launch_bounds__(128) void logits_gemm(const unsigned short* __restrict__ o_bf,
                                                   const float* __restrict__ R,
                                                   float* __restrict__ out) {
    __shared__ __align__(16) unsigned short o_s[32 * 128];
    const int tid = threadIdx.x;
    const int lane = tid & 31, wid = tid >> 5;
    const int half = lane >> 4, l15 = lane & 15;
    for (int idx = tid; idx < 2048; idx += 128)
        ((unsigned int*)o_s)[idx] = ((const unsigned int*)o_bf)[idx];
    __syncthreads();
    for (int tt = 0; tt < 4; ++tt) {
        const int tile = blockIdx.x * 16 + wid * 4 + tt;
        if (tile >= 2500) break;
        const int col = tile * 16 + l15;
        v8f acc0 = {}, acc1 = {};
        #pragma unroll
        for (int ks = 0; ks < 4; ++ks) {
            const int k0 = ks * 32;
            union { v16bf v; unsigned int u32[8]; } ub;
            const int kb0 = k0 + half * 16;
            #pragma unroll
            for (int kk = 0; kk < 8; ++kk) {
                float x0 = R[(size_t)(kb0 + 2 * kk) * 40000 + col];
                float x1 = R[(size_t)(kb0 + 2 * kk + 1) * 40000 + col];
                ub.u32[kk] = pack_bf16(x0, x1);
            }
            v16bf a0 = load_a_rows(o_s, l15, k0, half);
            acc0 = wmma_bf16(a0, ub.v, acc0);
            v16bf a1 = load_a_rows(o_s, 16 + l15, k0, half);
            acc1 = wmma_bf16(a1, ub.v, acc1);
        }
        #pragma unroll
        for (int r = 0; r < 8; ++r) {
            out[(size_t)(r + half * 8) * 40000 + col]      = acc0[r];
            out[(size_t)(16 + r + half * 8) * 40000 + col] = acc1[r];
        }
    }
}

// ================= launcher =================
extern "C" void kernel_launch(void* const* d_in, const int* in_sizes, int n_in,
                              void* d_out, int out_size, void* d_ws, size_t ws_size,
                              hipStream_t stream) {
    const float* memories = (const float*)d_in[0];
    const float* stories  = (const float*)d_in[1];
    const float* smask    = (const float*)d_in[2];
    const int*   queries  = (const int*)d_in[3];
    const int*   keys     = (const int*)d_in[4];
    const float* emb      = (const float*)d_in[5];
    const float* mmask    = (const float*)d_in[6];
    const float* A        = (const float*)d_in[7];
    const float* Bm       = (const float*)d_in[8];
    const float* C        = (const float*)d_in[9];
    const float* v        = (const float*)d_in[10];
    const float* D        = (const float*)d_in[11];
    const float* E        = (const float*)d_in[12];
    const float* F        = (const float*)d_in[13];
    const float* w        = (const float*)d_in[14];
    const float* H        = (const float*)d_in[15];
    const float* R        = (const float*)d_in[16];
    float* out = (float*)d_out;

    // workspace layout (bytes); needs ~8.3 MB (tm buffer dominates)
    char* ws = (char*)d_ws;
    float*          u_   = (float*)(ws + 0);
    float*          qc_  = (float*)(ws + 16384);
    float*          tf_  = (float*)(ws + 32768);
    float*          uH_  = (float*)(ws + 49152);
    float*          kb_  = (float*)(ws + 65536);
    float*          al2_ = (float*)(ws + 75776);
    unsigned short* ATbf = (unsigned short*)(ws + 141312);
    unsigned short* DTbf = (unsigned short*)(ws + 174080);
    unsigned short* ETbf = (unsigned short*)(ws + 206848);
    unsigned short* obf  = (unsigned short*)(ws + 239616);
    float*          tm_  = (float*)(ws + 247808);

    compute_u<<<32, 128, 0, stream>>>(queries, emb, mmask, u_);
    gemv_rows<<<20, 128, 0, stream>>>(nullptr, keys, emb, Bm, kb_);
    gemv_rows<<<32, 128, 0, stream>>>(u_, nullptr, nullptr, C, qc_);
    gemv_rows<<<32, 128, 0, stream>>>(u_, nullptr, nullptr, F, tf_);
    gemv_rows<<<32, 128, 0, stream>>>(u_, nullptr, nullptr, H, uH_);
    transpose_bf<<<128, 128, 0, stream>>>(A, ATbf);
    transpose_bf<<<128, 128, 0, stream>>>(D, DTbf);
    transpose_bf<<<128, 128, 0, stream>>>(E, ETbf);
    fused_intratemporal<<<dim3(32, 32), 128, 0, stream>>>(
        memories, stories, kb_, qc_, tf_, v, w, ATbf, DTbf, ETbf, tm_, al2_);
    finalize_o<<<32, 256, 0, stream>>>(al2_, smask, tm_, uH_, obf);
    logits_gemm<<<157, 128, 0, stream>>>(obf, R, out);
}
